// WindowAttention_83494164234928
// MI455X (gfx1250) — compile-verified, matrix operand order checked by reading, not run on hardware
//
#include <hip/hip_runtime.h>
#include <math.h>
#include <stdint.h>

typedef _Float16 half_t;
typedef __attribute__((ext_vector_type(16))) _Float16 v16h;
typedef __attribute__((ext_vector_type(8)))  _Float16 v8h;
typedef __attribute__((ext_vector_type(8)))  float    v8f;
typedef __attribute__((ext_vector_type(4)))  float    v4f;
typedef __attribute__((ext_vector_type(4)))  unsigned int v4u;
typedef __attribute__((ext_vector_type(8)))  int      v8i;
typedef __attribute__((ext_vector_type(4)))  int      v4i;

#define GK 512            // inner K of both big GEMMs
#define HEADS 16
#define NTOK 64
#define BW 1024

#if defined(__has_builtin)
#if __has_builtin(__builtin_amdgcn_tensor_load_to_lds) && __has_builtin(__builtin_amdgcn_s_wait_tensorcnt)
#define HAVE_TDM 1
#endif
#endif
#ifndef HAVE_TDM
#define HAVE_TDM 0
#endif

// ---- workspace layout (bytes) ----
static const size_t OFF_QKV    = 0;                       // f16 [65536][1536]  = 201326592
static const size_t OFF_AO     = 201326592;               // f16 [65536][512]   = 67108864
static const size_t OFF_RPB    = 268435456;               // f32 [16][64][64]   = 262144
static const size_t OFF_QKVWT  = 268697600;               // f16 [1536][512]    = 1572864
static const size_t OFF_PROJWT = 270270464;               // f16 [512][512]     = 524288
static const size_t OFF_QKVB   = 270794752;               // f32 [1536]

// ---------------------------------------------------------------------------
// WMMA fragment load from row-major f16 LDS tile (ISA 7.12.2 A/B layout).
// ---------------------------------------------------------------------------
__device__ __forceinline__ v16h frag_ld(const half_t* base, int row0, int k0, int ldk) {
    const int lane = threadIdx.x & 31;
    const int r  = row0 + (lane & 15);
    const int kb = k0 + ((lane >> 4) << 3);
    const half_t* p = base + r * ldk + kb;
    v8h lo = *reinterpret_cast<const v8h*>(p);
    v8h hi = *reinterpret_cast<const v8h*>(p + 16);
    v16h out;
#pragma unroll
    for (int i = 0; i < 8; ++i) { out[i] = lo[i]; out[i + 8] = hi[i]; }
    return out;
}

// ---- 16 contiguous elements -> two v8h (with f32->f16 convert if needed) ----
__device__ __forceinline__ void load_row16(const float* p, v8h& h0, v8h& h1) {
    v4f a0 = *(const v4f*)(p);     v4f a1 = *(const v4f*)(p + 4);
    v4f a2 = *(const v4f*)(p + 8); v4f a3 = *(const v4f*)(p + 12);
#pragma unroll
    for (int i = 0; i < 4; ++i) {
        h0[i] = (half_t)a0[i]; h0[i + 4] = (half_t)a1[i];
        h1[i] = (half_t)a2[i]; h1[i + 4] = (half_t)a3[i];
    }
}
__device__ __forceinline__ void load_row16(const half_t* p, v8h& h0, v8h& h1) {
    h0 = *(const v8h*)(p); h1 = *(const v8h*)(p + 8);
}
__device__ __forceinline__ void store_c(float* C, size_t i, float v)  { C[i] = v; }
__device__ __forceinline__ void store_c(half_t* C, size_t i, float v) { C[i] = (half_t)v; }

#if HAVE_TDM
// ---------------------------------------------------------------------------
// Tensor Data Mover: 2D f16 tile (tile_w elems x tile_h rows) global -> LDS.
// Descriptor per CDNA5 ISA (D# group0 + group1, 2D tensor).
// Issue once per wave; TDM ignores EXEC. Drain with s_wait_tensorcnt.
// ---------------------------------------------------------------------------
__device__ __forceinline__ void tdm_load_2d_f16(unsigned lds_addr, const half_t* gsrc,
                                                unsigned tile_w, unsigned tile_h,
                                                unsigned long long row_stride_elems) {
    const unsigned long long ga = (unsigned long long)(uintptr_t)gsrc;
    v4u g0;
    g0[0] = 1u;                                              // count=1, user-mode D#
    g0[1] = lds_addr;                                        // lds_addr (bytes)
    g0[2] = (unsigned)(ga & 0xffffffffull);                  // global_addr[31:0]
    g0[3] = (unsigned)((ga >> 32) & 0x01ffffffull) | (2u << 30); // addr[56:32] | type=2
    // group1 packed as 4x u64 (fields are not dword aligned)
    const unsigned long long q0 = 0x10000ull                                   // data_size=1 -> 2B
                                | ((unsigned long long)(tile_w & 0xffffu) << 48); // tensor_dim0 lo16
    const unsigned long long q1 = (unsigned long long)((tile_w >> 16) & 0xffffu)  // tensor_dim0 hi16
                                | ((unsigned long long)tile_h << 16)              // tensor_dim1
                                | ((unsigned long long)(tile_w & 0xffffu) << 48); // tile_dim0
    const unsigned long long q2 = (unsigned long long)(tile_h & 0xffffu)          // tile_dim1 (tile_dim2=0)
                                | ((row_stride_elems & 0xffffffffull) << 32);     // dim0_stride lo32
    const unsigned long long q3 = (row_stride_elems >> 32) & 0xffffull;           // dim0_stride hi16
    v8i g1;
    g1[0] = (int)(q0 & 0xffffffffull); g1[1] = (int)(q0 >> 32);
    g1[2] = (int)(q1 & 0xffffffffull); g1[3] = (int)(q1 >> 32);
    g1[4] = (int)(q2 & 0xffffffffull); g1[5] = (int)(q2 >> 32);
    g1[6] = (int)(q3 & 0xffffffffull); g1[7] = (int)(q3 >> 32);
    v4i z4 = {0, 0, 0, 0};
#if defined(__clang_major__) && (__clang_major__ >= 23)
    v8i z8 = {0, 0, 0, 0, 0, 0, 0, 0};
    __builtin_amdgcn_tensor_load_to_lds(g0, g1, z4, z4, z8, 0);
#else
    __builtin_amdgcn_tensor_load_to_lds(g0, g1, z4, z4, 0);
#endif
}
#endif // HAVE_TDM

// ---------------------------------------------------------------------------
// Tiled GEMM: C[M][ldc] = A[M][512] * BT[ldc][512]^T + bias
// 128x128 block tile, 8 waves, each wave 32x64 (2x4 fragments), K-step 32.
// ---------------------------------------------------------------------------
template <typename AT, typename CT>
__global__ __launch_bounds__(256)
void gemm_kernel(const AT* __restrict__ A, const half_t* __restrict__ BT,
                 const float* __restrict__ bias, CT* __restrict__ C, int ldc) {
    __shared__ half_t As[128][32];
    __shared__ half_t Bs[128][32];

    const int tid   = threadIdx.x;
    const int lane  = tid & 31;
    const int w     = tid >> 5;      // 0..7
    const int waveM = w >> 1;        // 0..3  (rows of 32)
    const int waveN = w & 1;         // 0..1  (cols of 64)
    const int m0 = blockIdx.x * 128;
    const int n0 = blockIdx.y * 128;

    const int lrow = tid >> 1;          // 0..127
    const int lcol = (tid & 1) << 4;    // 0 or 16

    v8f acc[2][4];
#pragma unroll
    for (int i = 0; i < 2; ++i)
#pragma unroll
        for (int j = 0; j < 4; ++j) acc[i][j] = {};

    for (int k0 = 0; k0 < GK; k0 += 32) {
        const AT*     ap = A  + (size_t)(m0 + lrow) * GK + k0 + lcol;
        const half_t* bp = BT + (size_t)(n0 + lrow) * GK + k0 + lcol;
        v8h a0, a1, b0, b1;
        load_row16(ap, a0, a1);
        load_row16(bp, b0, b1);
        *(v8h*)&As[lrow][lcol]     = a0;
        *(v8h*)&As[lrow][lcol + 8] = a1;
        *(v8h*)&Bs[lrow][lcol]     = b0;
        *(v8h*)&Bs[lrow][lcol + 8] = b1;
        if (k0 + 32 < GK) {                       // -> global_prefetch_b8
            __builtin_prefetch(ap + 32, 0, 1);
            __builtin_prefetch(bp + 32, 0, 1);
        }
        __syncthreads();

        v16h af[2], bf[4];
#pragma unroll
        for (int i = 0; i < 2; ++i) af[i] = frag_ld(&As[0][0], waveM * 32 + i * 16, 0, 32);
#pragma unroll
        for (int j = 0; j < 4; ++j) bf[j] = frag_ld(&Bs[0][0], waveN * 64 + j * 16, 0, 32);
#pragma unroll
        for (int i = 0; i < 2; ++i)
#pragma unroll
            for (int j = 0; j < 4; ++j)
                acc[i][j] = __builtin_amdgcn_wmma_f32_16x16x32_f16(
                    false, af[i], false, bf[j], (short)0, acc[i][j], false, false);
        __syncthreads();
    }

    const int lh = lane >> 4;     // C/D layout: row = r + 8*lh, col = lane&15
    const int lc = lane & 15;
#pragma unroll
    for (int i = 0; i < 2; ++i)
#pragma unroll
        for (int j = 0; j < 4; ++j) {
            const int mb = m0 + waveM * 32 + i * 16 + lh * 8;
            const int n  = n0 + waveN * 64 + j * 16 + lc;
            const float bv = bias[n];
#pragma unroll
            for (int r = 0; r < 8; ++r)
                store_c(C, (size_t)(mb + r) * ldc + n, acc[i][j][r] + bv);
        }
}

// ---------------------------------------------------------------------------
// Fused per-(window, head) cosine attention. Block = 128 threads (4 waves).
// q/k tiles staged LDS via Tensor Data Mover when available.
// ---------------------------------------------------------------------------
__global__ __launch_bounds__(128)
void attn_kernel(const half_t* __restrict__ qkv, const float* __restrict__ mask,
                 const float* __restrict__ rpb, const float* __restrict__ logit_scale,
                 half_t* __restrict__ out) {
    const int b = blockIdx.x;
    const int h = blockIdx.y;

    __shared__ half_t qs[64][32];
    __shared__ half_t ks2[64][32];
    __shared__ half_t vT[32][64];     // v transposed: vT[d][m]
    __shared__ float  S[64][64];
    __shared__ half_t P[64][64];

    const int tid  = threadIdx.x;
    const int lane = tid & 31;
    const int w    = tid >> 5;        // wave 0..3 -> 16 score rows each

    const size_t rowbase = (size_t)b * 64 * 1536 + (size_t)h * 32;

#if HAVE_TDM
    if (tid < 32) {   // wave 0 issues the TDM tile copies (EXEC-independent, per-wave)
        const unsigned q_lds = (unsigned)(uintptr_t)(void*)&qs[0][0];
        const unsigned k_lds = (unsigned)(uintptr_t)(void*)&ks2[0][0];
        tdm_load_2d_f16(q_lds, qkv + rowbase,        32, 64, 1536ull);
        tdm_load_2d_f16(k_lds, qkv + rowbase + 512,  32, 64, 1536ull);
        __builtin_amdgcn_s_wait_tensorcnt(0);
    }
#else
    { // load q (threads 0..63) / k (threads 64..127), 32 halves each
        const int r   = tid & 63;
        const int sel = tid >> 6;                       // 0=q, 1=k
        const half_t* src = qkv + rowbase + (size_t)r * 1536 + sel * 512;
        v8h x0 = *(const v8h*)(src);      v8h x1 = *(const v8h*)(src + 8);
        v8h x2 = *(const v8h*)(src + 16); v8h x3 = *(const v8h*)(src + 24);
        half_t* dst = sel ? &ks2[r][0] : &qs[r][0];
        *(v8h*)(dst) = x0; *(v8h*)(dst + 8) = x1; *(v8h*)(dst + 16) = x2; *(v8h*)(dst + 24) = x3;
    }
#endif
    if (tid < 64) { // v, stored transposed for the B-operand
        const half_t* src = qkv + rowbase + (size_t)tid * 1536 + 1024;
#pragma unroll
        for (int d = 0; d < 32; ++d) vT[d][tid] = src[d];
    }
    __syncthreads();

    { // L2-normalize q and k rows; fold per-head logit scale into q
        const int r = tid & 63;
        half_t* row = (tid < 64) ? &qs[r][0] : &ks2[r][0];
        float ss = 0.f;
#pragma unroll
        for (int d = 0; d < 32; ++d) { float x = (float)row[d]; ss += x * x; }
        float inv = rsqrtf(fmaxf(ss, 1e-12f));
        if (tid < 64) inv *= expf(fminf(logit_scale[h], 4.6051702f));   // ln(100)
#pragma unroll
        for (int d = 0; d < 32; ++d) row[d] = (half_t)((float)row[d] * inv);
    }
    __syncthreads();

    { // S = qn * kn^T  (64x64, K=32) + rpb + mask
        const float* maskp = mask + (size_t)(b & 63) * 4096;
        const float* rpbp  = rpb  + (size_t)h * 4096;
        v16h aq = frag_ld(&qs[0][0], w * 16, 0, 32);
#pragma unroll
        for (int j = 0; j < 4; ++j) {
            v16h bk = frag_ld(&ks2[0][0], j * 16, 0, 32);
            v8f acc = {};
            acc = __builtin_amdgcn_wmma_f32_16x16x32_f16(
                false, aq, false, bk, (short)0, acc, false, false);
            const int mrow = w * 16 + ((lane >> 4) << 3);
            const int col  = j * 16 + (lane & 15);
#pragma unroll
            for (int r = 0; r < 8; ++r) {
                const int m = mrow + r;
                S[m][col] = acc[r] + rpbp[m * 64 + col] + maskp[m * 64 + col];
            }
        }
    }
    __syncthreads();

    if (tid < 64) { // row softmax -> f16 probs
        float mx = -1e30f;
#pragma unroll
        for (int c = 0; c < 64; ++c) mx = fmaxf(mx, S[tid][c]);
        float sum = 0.f;
#pragma unroll
        for (int c = 0; c < 64; ++c) { float e = expf(S[tid][c] - mx); S[tid][c] = e; sum += e; }
        const float inv = 1.f / sum;
#pragma unroll
        for (int c = 0; c < 64; ++c) P[tid][c] = (half_t)(S[tid][c] * inv);
    }
    __syncthreads();

    { // O = P @ V  (64x32, K=64 in two WMMA steps)
        v8f acc0 = {}, acc1 = {};
#pragma unroll
        for (int kk = 0; kk < 2; ++kk) {
            v16h ap = frag_ld(&P[0][0], w * 16, kk * 32, 64);
            v16h b0 = frag_ld(&vT[0][0], 0,  kk * 32, 64);
            v16h b1 = frag_ld(&vT[0][0], 16, kk * 32, 64);
            acc0 = __builtin_amdgcn_wmma_f32_16x16x32_f16(false, ap, false, b0, (short)0, acc0, false, false);
            acc1 = __builtin_amdgcn_wmma_f32_16x16x32_f16(false, ap, false, b1, (short)0, acc1, false, false);
        }
        const int mrow = w * 16 + ((lane >> 4) << 3);
        const int col  = lane & 15;
#pragma unroll
        for (int r = 0; r < 8; ++r) {
            const size_t base = ((size_t)b * 64 + mrow + r) * 512 + h * 32;
            out[base + col]      = (half_t)acc0[r];
            out[base + 16 + col] = (half_t)acc1[r];
        }
    }
}

// ---------------------------------------------------------------------------
// Prep kernels
// ---------------------------------------------------------------------------
__global__ void transpose_f32_to_f16(const float* __restrict__ W, half_t* __restrict__ WT,
                                     int K, int Ncols) {   // W:[K][N] -> WT:[N][K]
    int idx = blockIdx.x * blockDim.x + threadIdx.x;
    if (idx >= K * Ncols) return;
    int n = idx / K, k = idx % K;
    WT[(size_t)n * K + k] = (half_t)W[(size_t)k * Ncols + n];
}

__global__ void build_qkv_bias(const float* __restrict__ qb, const float* __restrict__ vb,
                               float* __restrict__ outb) {
    int i = blockIdx.x * blockDim.x + threadIdx.x;
    if (i >= 1536) return;
    float v = 0.f;
    if (i < 512) v = qb[i];
    else if (i >= 1024) v = vb[i - 1024];
    outb[i] = v;
}

__device__ __forceinline__ float cpb_coord(int idx) {     // Swin-V2 log-CPB coord
    float t = (float)(idx - 7) * (8.0f / 7.0f);
    float s = (t > 0.f) ? 1.f : ((t < 0.f) ? -1.f : 0.f);
    return s * log2f(fabsf(t) + 1.f) * (1.0f / 3.0f);     // /log2(8)
}

__global__ __launch_bounds__(256)
void cpb_kernel(const float* __restrict__ w1, const float* __restrict__ b1,
                const float* __restrict__ w2, float* __restrict__ rpb) {
    __shared__ float sw1[2][512];
    __shared__ float sb1[512];
    __shared__ float sw2[512][16];
    __shared__ float table[225][16];
    const int tid = threadIdx.x;
    for (int i = tid; i < 512; i += 256) {
        sw1[0][i] = w1[i]; sw1[1][i] = w1[512 + i]; sb1[i] = b1[i];
    }
    for (int i = tid; i < 512 * 16; i += 256) sw2[i >> 4][i & 15] = w2[i];
    __syncthreads();
    for (int e = tid; e < 225 * 16; e += 256) {     // MLP: relu(t*W1+b1)*W2
        const int p = e >> 4, hh = e & 15;
        const float t0 = cpb_coord(p % 15);         // xy meshgrid flatten
        const float t1 = cpb_coord(p / 15);
        float s = 0.f;
        for (int q = 0; q < 512; ++q) {
            float hv = fmaxf(t0 * sw1[0][q] + t1 * sw1[1][q] + sb1[q], 0.f);
            s += hv * sw2[q][hh];
        }
        table[p][hh] = s;
    }
    __syncthreads();
    for (int e = tid; e < HEADS * 64 * 64; e += 256) {  // gather + 16*sigmoid
        const int h = e >> 12, nn = (e >> 6) & 63, mm = e & 63;
        const int r0 = (nn & 7) - (mm & 7);
        const int r1 = (nn >> 3) - (mm >> 3);
        const float v = table[(r0 + 7) * 15 + (r1 + 7)][h];
        rpb[e] = 16.f / (1.f + expf(-v));
    }
}

// ---------------------------------------------------------------------------
extern "C" void kernel_launch(void* const* d_in, const int* in_sizes, int n_in,
                              void* d_out, int out_size, void* d_ws, size_t ws_size,
                              hipStream_t stream) {
    const float* x           = (const float*)d_in[0];
    const float* mask        = (const float*)d_in[1];
    const float* qkv_w       = (const float*)d_in[2];
    const float* q_bias      = (const float*)d_in[3];
    const float* v_bias      = (const float*)d_in[4];
    const float* logit_scale = (const float*)d_in[5];
    const float* cpb_w1      = (const float*)d_in[6];
    const float* cpb_b1      = (const float*)d_in[7];
    const float* cpb_w2      = (const float*)d_in[8];
    const float* proj_w      = (const float*)d_in[9];
    const float* proj_b      = (const float*)d_in[10];

    char* ws = (char*)d_ws;
    half_t* qkv16  = (half_t*)(ws + OFF_QKV);
    half_t* ao16   = (half_t*)(ws + OFF_AO);
    float*  rpb    = (float*) (ws + OFF_RPB);
    half_t* qkvwT  = (half_t*)(ws + OFF_QKVWT);
    half_t* projwT = (half_t*)(ws + OFF_PROJWT);
    float*  qkvb   = (float*) (ws + OFF_QKVB);

    transpose_f32_to_f16<<<(512 * 1536 + 255) / 256, 256, 0, stream>>>(qkv_w, qkvwT, 512, 1536);
    transpose_f32_to_f16<<<(512 * 512 + 255) / 256, 256, 0, stream>>>(proj_w, projwT, 512, 512);
    build_qkv_bias<<<6, 256, 0, stream>>>(q_bias, v_bias, qkvb);
    cpb_kernel<<<1, 256, 0, stream>>>(cpb_w1, cpb_b1, cpb_w2, rpb);

    // QKV: [65536 x 512] * [512 x 1536] -> f16 ws
    gemm_kernel<float, half_t><<<dim3(512, 12), 256, 0, stream>>>(x, qkvwT, qkvb, qkv16, 1536);
    // fused window attention
    attn_kernel<<<dim3(BW, HEADS), 128, 0, stream>>>(qkv16, mask, rpb, logit_scale, ao16);
    // Proj: [65536 x 512] * [512 x 512] -> f32 d_out
    gemm_kernel<half_t, float><<<dim3(512, 4), 256, 0, stream>>>(ao16, projwT, proj_b, (float*)d_out, 512);
}
